// SAGEClassifier_69999376990327
// MI455X (gfx1250) — matrix operand order repaired and checked by reading, not compiled, not run on hardware
//
#include <hip/hip_runtime.h>
#include <hip/hip_bf16.h>

// ---------------------------------------------------------------------------
// Problem constants (from reference)
#define NNODES   50000
#define DEG      16
#define IN_DIM   256
#define HID      128
#define NGRAPHS  256
#define NCLS     10

typedef __bf16 bf16;
typedef bf16  v16bf __attribute__((ext_vector_type(16)));
typedef float v8f   __attribute__((ext_vector_type(8)));

union V16U { v16bf v; uint4 q[2]; };

// ---------------------------------------------------------------------------
// WMMA tile loaders (bf16, 16x16x32, wave32 layouts per CDNA5 ISA 7.12.2)
//
// A (16x32, M x K): lane l (0..15) holds row M=l, K=0..7 and K=16..23;
//                   lane l+16 holds row M=l, K=8..15 and K=24..31.
__device__ __forceinline__ v16bf load_tile_a(const bf16* __restrict__ p0, int ld) {
    int lane  = threadIdx.x & 31;
    int m     = lane & 15;
    int kbase = (lane >> 4) ? 8 : 0;
    const bf16* p = p0 + (size_t)m * ld + kbase;
    V16U u;
    u.q[0] = *(const uint4*)(p);        // K = kbase .. kbase+7
    u.q[1] = *(const uint4*)(p + 16);   // K = kbase+16 .. kbase+23
    return u.v;
}

// B (32x16, K x N): column n of B == row n of W (row-major -> contiguous).
// lane l (0..15) holds column N=l, K=0..7 and K=16..23; lane l+16 holds column N=l, K=8..15 and K=24..31.
// Works for both global and LDS-resident panels (addrspace inferred after inline).
__device__ __forceinline__ v16bf load_tile_b(const bf16* p0, int ld) {
    int lane  = threadIdx.x & 31;
    int n     = lane & 15;
    int kbase = (lane >> 4) ? 8 : 0;
    const bf16* p = p0 + (size_t)n * ld + kbase;
    V16U u;
    u.q[0] = *(const uint4*)(p);
    u.q[1] = *(const uint4*)(p + 16);
    return u.v;
}

__device__ __forceinline__ v8f wmma_bf16(v16bf a, v16bf b, v8f c) {
    return __builtin_amdgcn_wmma_f32_16x16x32_bf16(false, a, false, b, (short)0, c,
                                                   false, false);
}

__device__ __forceinline__ float sigmoidf_fast(float x) {
    return 1.0f / (1.0f + __expf(-x));
}

// ---------------------------------------------------------------------------
// Elementwise helpers
__global__ void f32_to_bf16_kernel(const float* __restrict__ src,
                                   bf16* __restrict__ dst, int n) {
    int i = blockIdx.x * blockDim.x + threadIdx.x;
    if (i < n) dst[i] = (bf16)src[i];
}

__global__ void vec_add_kernel(const float* __restrict__ a,
                               const float* __restrict__ b,
                               float* __restrict__ o, int n) {
    int i = blockIdx.x * blockDim.x + threadIdx.x;
    if (i < n) o[i] = a[i] + b[i];
}

// ---------------------------------------------------------------------------
// C[M,Nout](bf16) = A[M,K](bf16) @ W[Nout,K]^T + bias(f32)
// block = 128 threads = 4 waves; each wave owns one 16x16 C tile.
// The 16-column W panel is staged in LDS once per block (shared by 4 waves).
template <int K>
__global__ __launch_bounds__(128)
void gemm_bias_bf16_kernel(const bf16* __restrict__ A,
                           const bf16* __restrict__ W,
                           const float* __restrict__ bias,
                           bf16* __restrict__ C,
                           int M, int Nout) {
    __shared__ __align__(16) bf16 Bs[16 * K];
    constexpr int CH = K / 8;               // uint4 chunks per row
    int colBase = blockIdx.y << 4;
    for (int i = threadIdx.x; i < 16 * CH; i += 128) {
        int row = i / CH, ch = i - row * CH;
        ((uint4*)Bs)[i] = *(const uint4*)(W + (size_t)(colBase + row) * K + ch * 8);
    }
    __syncthreads();

    int wave    = threadIdx.x >> 5;
    int lane    = threadIdx.x & 31;
    int rowTile = blockIdx.x * 4 + wave;
    int Mtiles  = M >> 4;
    if (rowTile >= Mtiles) return;          // wave-uniform: EXEC all-1s for WMMA

    const bf16* Abase = A + (size_t)rowTile * 16 * K;

    v8f acc = {};
    for (int k = 0; k < K; k += 32) {
        v16bf a = load_tile_a(Abase + k, K);
        v16bf b = load_tile_b(Bs + k, K);   // ds_load_b128 path
        acc = wmma_bf16(a, b, acc);
    }

    int n     = lane & 15;
    int mhalf = (lane >> 4) * 8;
    float bv  = bias ? bias[colBase + n] : 0.0f;
    for (int r = 0; r < 8; ++r) {
        int row = rowTile * 16 + mhalf + r;
        C[(size_t)row * Nout + colBase + n] = (bf16)(acc[r] + bv);
    }
}

// ---------------------------------------------------------------------------
// One LSTM timestep, fully fused:
//   gates = gather(P, idx[:,t]) + h_prev @ W_hh^T   (P already holds x@W_ih^T + b)
//   c = sig(f)*c + sig(i)*tanh(g);  h_out = sig(o)*tanh(c)
// Each wave: 16 nodes x 16 hidden units; 4 accumulators (i,f,g,o) share the A
// tile; all 4 waves share the 4-gate W_hh panel staged in LDS (32KB @ D=256).
template <int D>
__global__ __launch_bounds__(128)
void lstm_step_kernel(const bf16* __restrict__ P,      // [Np, 4D] (i,f,g,o blocks)
                      const int*  __restrict__ nidx,   // [Np, DEG]
                      int t,
                      const bf16* __restrict__ Whh,    // [4D, D]
                      const bf16* __restrict__ hprev,  // [Np, D]
                      bf16*       __restrict__ hout,   // [Np, D]
                      float*      __restrict__ c,      // [Np, D]
                      int Np) {
    __shared__ __align__(16) bf16 Bs[4 * 16 * D];
    constexpr int CH = D / 8;               // uint4 chunks per row
    int jBase = blockIdx.y << 4;
    for (int i = threadIdx.x; i < 64 * CH; i += 128) {
        int row = i / CH, ch = i - row * CH;    // row = g*16 + n
        int g = row >> 4, n = row & 15;
        ((uint4*)Bs)[i] =
            *(const uint4*)(Whh + (size_t)(g * D + jBase + n) * D + ch * 8);
    }
    __syncthreads();

    int wave    = threadIdx.x >> 5;
    int lane    = threadIdx.x & 31;
    int rowTile = blockIdx.x * 4 + wave;
    int Mtiles  = Np >> 4;
    if (rowTile >= Mtiles) return;
    constexpr int fourD = D * 4;
    int n     = lane & 15;
    int mhalf = (lane >> 4) * 8;

    // Init accumulators from gathered precomputed input projections (C-layout).
    v8f acc_i = {}, acc_f = {}, acc_g = {}, acc_o = {};
    for (int r = 0; r < 8; ++r) {
        int node = rowTile * 16 + mhalf + r;
        int src  = nidx[node * DEG + t];
        const bf16* prow = P + (size_t)src * fourD + jBase + n;
        acc_i[r] = (float)prow[0];
        acc_f[r] = (float)prow[D];
        acc_g[r] = (float)prow[2 * D];
        acc_o[r] = (float)prow[3 * D];
    }

    // Recurrent GEMM: one A tile feeds four WMMAs (i,f,g,o gate columns).
    const bf16* Abase = hprev + (size_t)rowTile * 16 * D;
    for (int k = 0; k < D; k += 32) {
        v16bf a = load_tile_a(Abase + k, D);
        acc_i = wmma_bf16(a, load_tile_b(Bs + 0 * 16 * D + k, D), acc_i);
        acc_f = wmma_bf16(a, load_tile_b(Bs + 1 * 16 * D + k, D), acc_f);
        acc_g = wmma_bf16(a, load_tile_b(Bs + 2 * 16 * D + k, D), acc_g);
        acc_o = wmma_bf16(a, load_tile_b(Bs + 3 * 16 * D + k, D), acc_o);
    }

    // Fused cell update.
    for (int r = 0; r < 8; ++r) {
        int node  = rowTile * 16 + mhalf + r;
        size_t of = (size_t)node * D + jBase + n;
        float ig = sigmoidf_fast(acc_i[r]);
        float fg = sigmoidf_fast(acc_f[r]);
        float og = sigmoidf_fast(acc_o[r]);
        float gg = tanhf(acc_g[r]);
        float cn = fg * c[of] + ig * gg;
        c[of]    = cn;
        hout[of] = (bf16)(og * tanhf(cn));
    }
}

// ---------------------------------------------------------------------------
// out = relu(A1 @ Wself^T + A2 @ Wneigh^T + b); writes bf16 and/or f32.
// Both 16-column weight panels staged in LDS.
template <int K1, int K2>
__global__ __launch_bounds__(128)
void fc_relu_kernel(const bf16* __restrict__ A1,
                    const bf16* __restrict__ Wself,
                    const bf16* __restrict__ A2,
                    const bf16* __restrict__ Wneigh,
                    const float* __restrict__ bias,
                    bf16*  __restrict__ out_bf,
                    float* __restrict__ out_f32,
                    int M, int Nout) {
    __shared__ __align__(16) bf16 Bs[16 * (K1 + K2)];
    constexpr int CH1 = K1 / 8, CH2 = K2 / 8;
    int colBase = blockIdx.y << 4;
    for (int i = threadIdx.x; i < 16 * CH1; i += 128) {
        int row = i / CH1, ch = i - row * CH1;
        ((uint4*)Bs)[i] = *(const uint4*)(Wself + (size_t)(colBase + row) * K1 + ch * 8);
    }
    uint4* Bs2q = (uint4*)(Bs + 16 * K1);
    for (int i = threadIdx.x; i < 16 * CH2; i += 128) {
        int row = i / CH2, ch = i - row * CH2;
        Bs2q[i] = *(const uint4*)(Wneigh + (size_t)(colBase + row) * K2 + ch * 8);
    }
    __syncthreads();

    int wave    = threadIdx.x >> 5;
    int lane    = threadIdx.x & 31;
    int rowTile = blockIdx.x * 4 + wave;
    int Mtiles  = M >> 4;
    if (rowTile >= Mtiles) return;

    v8f acc = {};
    {
        const bf16* Abase = A1 + (size_t)rowTile * 16 * K1;
        for (int k = 0; k < K1; k += 32)
            acc = wmma_bf16(load_tile_a(Abase + k, K1), load_tile_b(Bs + k, K1), acc);
    }
    {
        const bf16* Abase = A2 + (size_t)rowTile * 16 * K2;
        const bf16* B2 = Bs + 16 * K1;
        for (int k = 0; k < K2; k += 32)
            acc = wmma_bf16(load_tile_a(Abase + k, K2), load_tile_b(B2 + k, K2), acc);
    }

    int n     = lane & 15;
    int mhalf = (lane >> 4) * 8;
    float bv  = bias[colBase + n];
    for (int r = 0; r < 8; ++r) {
        int row = rowTile * 16 + mhalf + r;
        float v = acc[r] + bv;
        v = v > 0.0f ? v : 0.0f;
        size_t of = (size_t)row * Nout + colBase + n;
        if (out_bf)  out_bf[of]  = (bf16)v;
        if (out_f32) out_f32[of] = v;
    }
}

// ---------------------------------------------------------------------------
// Per-graph max (values are post-ReLU >= 0 -> int-bits atomicMax valid; hg init 0).
__global__ void segmax_kernel(const float* __restrict__ h2,
                              const int* __restrict__ gid,
                              float* __restrict__ hg, int Np, int H) {
    int i = blockIdx.x * blockDim.x + threadIdx.x;
    if (i >= Np * H) return;
    int node = i / H;
    int j    = i - node * H;
    int g    = gid[node];
    atomicMax((int*)&hg[g * H + j], __float_as_int(h2[i]));
}

// Classifier: out[G, C] = hg @ Wc^T + bc (tiny; plain VALU).
__global__ void cls_kernel(const float* __restrict__ hg,
                           const float* __restrict__ Wc,
                           const float* __restrict__ bc,
                           float* __restrict__ out, int G, int H, int C) {
    int i = blockIdx.x * blockDim.x + threadIdx.x;
    if (i >= G * C) return;
    int g = i / C, cc = i - g * C;
    float s = bc[cc];
    const float* hr = hg + (size_t)g * H;
    const float* wr = Wc + (size_t)cc * H;
    for (int k = 0; k < H; ++k) s += hr[k] * wr[k];
    out[i] = s;
}

// ---------------------------------------------------------------------------
extern "C" void kernel_launch(void* const* d_in, const int* in_sizes, int n_in,
                              void* d_out, int out_size, void* d_ws, size_t ws_size,
                              hipStream_t stream) {
    const float* h_in   = (const float*)d_in[0];
    const int*   nidx   = (const int*)d_in[1];
    const int*   gids   = (const int*)d_in[2];
    const float* W_ih1  = (const float*)d_in[3];
    const float* W_hh1  = (const float*)d_in[4];
    const float* b_ih1  = (const float*)d_in[5];
    const float* b_hh1  = (const float*)d_in[6];
    const float* W_sf1  = (const float*)d_in[7];
    const float* W_ng1  = (const float*)d_in[8];
    const float* b1     = (const float*)d_in[9];
    const float* W_ih2  = (const float*)d_in[10];
    const float* W_hh2  = (const float*)d_in[11];
    const float* b_ih2  = (const float*)d_in[12];
    const float* b_hh2  = (const float*)d_in[13];
    const float* W_sf2  = (const float*)d_in[14];
    const float* W_ng2  = (const float*)d_in[15];
    const float* b2     = (const float*)d_in[16];
    const float* W_cls  = (const float*)d_in[17];
    const float* b_cls  = (const float*)d_in[18];
    float* out = (float*)d_out;

    const int N = NNODES, D1 = IN_DIM, H = HID;

    // --- bump allocator over d_ws (256B aligned) ---
    char* base = (char*)d_ws;
    size_t off = 0;
    auto alloc = [&](size_t bytes) -> void* {
        void* p = base + off;
        off += (bytes + 255) & ~(size_t)255;
        return p;
    };

    // persistent region
    bf16* hbf     = (bf16*)alloc((size_t)N * D1 * 2);
    bf16* Wih1b   = (bf16*)alloc((size_t)4 * D1 * D1 * 2);
    bf16* Whh1b   = (bf16*)alloc((size_t)4 * D1 * D1 * 2);
    bf16* Wsf1b   = (bf16*)alloc((size_t)H * D1 * 2);
    bf16* Wng1b   = (bf16*)alloc((size_t)H * D1 * 2);
    bf16* Wih2b   = (bf16*)alloc((size_t)4 * H * H * 2);
    bf16* Whh2b   = (bf16*)alloc((size_t)4 * H * H * 2);
    bf16* Wsf2b   = (bf16*)alloc((size_t)H * H * 2);
    bf16* Wng2b   = (bf16*)alloc((size_t)H * H * 2);
    float* bsum1  = (float*)alloc((size_t)4 * D1 * 4);
    float* bsum2  = (float*)alloc((size_t)4 * H * 4);
    bf16* h1bf    = (bf16*)alloc((size_t)N * H * 2);
    float* h2f    = (float*)alloc((size_t)N * H * 4);
    float* hg     = (float*)alloc((size_t)NGRAPHS * H * 4);

    size_t scratchMark = off;
    // conv1 scratch
    bf16* P1  = (bf16*)alloc((size_t)N * 4 * D1 * 2);
    bf16* hA1 = (bf16*)alloc((size_t)N * D1 * 2);
    bf16* hB1 = (bf16*)alloc((size_t)N * D1 * 2);
    float* c1 = (float*)alloc((size_t)N * D1 * 4);
    // conv2 scratch overlays conv1 scratch
    off = scratchMark;
    bf16* P2  = (bf16*)alloc((size_t)N * 4 * H * 2);
    bf16* hA2 = (bf16*)alloc((size_t)N * H * 2);
    bf16* hB2 = (bf16*)alloc((size_t)N * H * 2);
    float* c2 = (float*)alloc((size_t)N * H * 4);
    (void)ws_size; (void)n_in; (void)in_sizes; (void)out_size;

    const int Mtiles = N / 16;                 // 3125 exactly
    const int gx     = (Mtiles + 3) / 4;       // 4 waves per block

    auto cvt = [&](const float* s, bf16* d, int n) {
        f32_to_bf16_kernel<<<(n + 255) / 256, 256, 0, stream>>>(s, d, n);
    };

    // ---- precision conversion + bias folding ----
    cvt(h_in,  hbf,   N * D1);
    cvt(W_ih1, Wih1b, 4 * D1 * D1);
    cvt(W_hh1, Whh1b, 4 * D1 * D1);
    cvt(W_sf1, Wsf1b, H * D1);
    cvt(W_ng1, Wng1b, H * D1);
    cvt(W_ih2, Wih2b, 4 * H * H);
    cvt(W_hh2, Whh2b, 4 * H * H);
    cvt(W_sf2, Wsf2b, H * H);
    cvt(W_ng2, Wng2b, H * H);
    vec_add_kernel<<<(4 * D1 + 255) / 256, 256, 0, stream>>>(b_ih1, b_hh1, bsum1, 4 * D1);
    vec_add_kernel<<<(4 * H + 255) / 256, 256, 0, stream>>>(b_ih2, b_hh2, bsum2, 4 * H);

    // ================= conv1 =================
    // P1 = h @ W_ih1^T + (b_ih1 + b_hh1)   [N, 1024]
    gemm_bias_bf16_kernel<IN_DIM><<<dim3(gx, (4 * D1) / 16), 128, 0, stream>>>(
        hbf, Wih1b, bsum1, P1, N, 4 * D1);

    hipMemsetAsync(hA1, 0, (size_t)N * D1 * 2, stream);
    hipMemsetAsync(c1,  0, (size_t)N * D1 * 4, stream);
    for (int t = 0; t < DEG; ++t) {
        const bf16* hp = (t & 1) ? hB1 : hA1;
        bf16*       ho = (t & 1) ? hA1 : hB1;
        lstm_step_kernel<IN_DIM><<<dim3(gx, D1 / 16), 128, 0, stream>>>(
            P1, nidx, t, Whh1b, hp, ho, c1, N);
    }
    // after t=15 (odd) final hidden is in hA1
    fc_relu_kernel<IN_DIM, IN_DIM><<<dim3(gx, H / 16), 128, 0, stream>>>(
        hbf, Wsf1b, hA1, Wng1b, b1, h1bf, (float*)nullptr, N, H);

    // ================= conv2 =================
    gemm_bias_bf16_kernel<HID><<<dim3(gx, (4 * H) / 16), 128, 0, stream>>>(
        h1bf, Wih2b, bsum2, P2, N, 4 * H);

    hipMemsetAsync(hA2, 0, (size_t)N * H * 2, stream);
    hipMemsetAsync(c2,  0, (size_t)N * H * 4, stream);
    for (int t = 0; t < DEG; ++t) {
        const bf16* hp = (t & 1) ? hB2 : hA2;
        bf16*       ho = (t & 1) ? hA2 : hB2;
        lstm_step_kernel<HID><<<dim3(gx, H / 16), 128, 0, stream>>>(
            P2, nidx, t, Whh2b, hp, ho, c2, N);
    }
    fc_relu_kernel<HID, HID><<<dim3(gx, H / 16), 128, 0, stream>>>(
        h1bf, Wsf2b, hA2, Wng2b, b2, (bf16*)nullptr, h2f, N, H);

    // ================= readout + classifier =================
    hipMemsetAsync(hg, 0, (size_t)NGRAPHS * H * 4, stream);
    segmax_kernel<<<((N * H) + 255) / 256, 256, 0, stream>>>(h2f, gids, hg, N, H);
    cls_kernel<<<(NGRAPHS * NCLS + 127) / 128, 128, 0, stream>>>(
        hg, W_cls, b_cls, out, NGRAPHS, H, NCLS);
}
